// Qwen2MoeDecoderLayer_51333449121908
// MI455X (gfx1250) — compile-verified, matrix operand order checked by reading, not run on hardware
//
#include <hip/hip_runtime.h>
#include <hip/hip_bf16.h>
#include <math.h>

#define DEVI __device__ __forceinline__

typedef _Float16 v8h  __attribute__((ext_vector_type(8)));
typedef _Float16 v16h __attribute__((ext_vector_type(16)));
typedef float    v8f  __attribute__((ext_vector_type(8)));

constexpr int   T_     = 1024;
constexpr int   HID_   = 2048;
constexpr int   HEADS_ = 16;
constexpr int   HD_    = 128;
constexpr int   E_     = 60;
constexpr int   TOPK_  = 4;
constexpr int   MI_    = 1408;
constexpr int   SI_    = 5632;
constexpr int   NPAIR_ = T_ * TOPK_;      // 4096 token-expert pairs
constexpr float EPS_   = 1e-6f;
constexpr float SCALE_ = 0.08838834764831845f;  // HD^-0.5

// ---------------------------------------------------------------------------
// WMMA helpers (gfx1250 wave32, 16x16x32 f16 -> f32)
// ---------------------------------------------------------------------------
DEVI v8f wmma16(v16h a, v16h b, v8f c) {
  return __builtin_amdgcn_wmma_f32_16x16x32_f16(
      /*neg_a=*/false, a, /*neg_b=*/false, b,
      /*c_mod=*/(short)0, c, /*reuse_a=*/false, /*reuse_b=*/false);
}

// A-fragment (16x32 f16): lane L holds row L%16; element e maps to
// K = (e/8)*16 + (L/16)*8 + e%8 -> two contiguous 8-halfword runs.
DEVI v16h load_a_frag(const _Float16* rowptr, int kk, int hi) {
  v8h lo = *(const v8h*)(rowptr + kk + hi * 8);
  v8h hh = *(const v8h*)(rowptr + kk + 16 + hi * 8);
  v16h a;
#pragma unroll
  for (int i = 0; i < 8; ++i) { a[i] = lo[i]; a[8 + i] = hh[i]; }
  return a;
}

// Four B-fragments (cols n..n+63, 32-deep k) from row-major fp32 [K x LDB].
// bptr = B + (hi*8)*LDB + n0 + lane%16.  All element offsets are compile-time
// immediates -> one address register, one load clause, one wait, then cvts.
template <int LDB>
DEVI void load_b4_f32(const float* __restrict__ bptr, v16h (&bf)[4]) {
  float t[4][16];
#pragma unroll
  for (int j = 0; j < 4; ++j)
#pragma unroll
    for (int e = 0; e < 16; ++e)
      t[j][e] = bptr[((e >> 3) * 16 + (e & 7)) * LDB + j * 16];
#pragma unroll
  for (int j = 0; j < 4; ++j)
#pragma unroll
    for (int e = 0; e < 16; ++e) bf[j][e] = (_Float16)t[j][e];
}

DEVI float silu_f(float g) { return g / (1.0f + __expf(-g)); }

// ---------------------------------------------------------------------------
// RMSNorm: fp32 in -> f16 out
// ---------------------------------------------------------------------------
__global__ void rmsnorm_kernel(const float* __restrict__ X,
                               const float* __restrict__ W,
                               _Float16* __restrict__ O) {
  int t = blockIdx.x, tid = threadIdx.x;
  __shared__ float red[8];
  __shared__ float invs;
  float ss = 0.0f;
  for (int c = tid; c < HID_; c += 256) {
    float v = X[(size_t)t * HID_ + c];
    ss += v * v;
  }
#pragma unroll
  for (int d = 1; d < 32; d <<= 1) ss += __shfl_xor(ss, d, 32);
  if ((tid & 31) == 0) red[tid >> 5] = ss;
  __syncthreads();
  if (tid == 0) {
    float s = 0.0f;
    for (int i = 0; i < 8; ++i) s += red[i];
    invs = rsqrtf(s / (float)HID_ + EPS_);
  }
  __syncthreads();
  float inv = invs;
  for (int c = tid; c < HID_; c += 256)
    O[(size_t)t * HID_ + c] = (_Float16)(X[(size_t)t * HID_ + c] * inv * W[c]);
}

// ---------------------------------------------------------------------------
// Dense WMMA GEMM: C[M,N](f32) = A_f16[M,lda] x B_f32[K,N] (+bias)(+residual)
// One wave per block; wave computes a 32(M) x 64(N) tile. grid = (M/32, N/64)
// ---------------------------------------------------------------------------
template <int N, int KDIM>
__global__ void __launch_bounds__(32, 1)
gemm_wmma_kernel(const _Float16* __restrict__ A, int lda,
                 const float* __restrict__ B,
                 const float* __restrict__ bias,
                 float* __restrict__ C,
                 const float* __restrict__ residual) {
  int lane = threadIdx.x, lm = lane & 15, hi = lane >> 4;
  int mbase = blockIdx.x * 32;
  int n0 = blockIdx.y * 64;
  const _Float16* aptr0 = A + (size_t)(mbase + lm) * lda;
  const _Float16* aptr1 = A + (size_t)(mbase + 16 + lm) * lda;
  const float* bptr = B + (size_t)(hi * 8) * N + n0 + lm;

  v8f acc0[4], acc1[4];
#pragma unroll
  for (int j = 0; j < 4; ++j) {
    float bv = bias ? bias[n0 + j * 16 + lm] : 0.0f;
#pragma unroll
    for (int v = 0; v < 8; ++v) { acc0[j][v] = bv; acc1[j][v] = bv; }
  }
  for (int kk = 0; kk < KDIM; kk += 32) {
    v16h a0 = load_a_frag(aptr0, 0, hi);
    v16h a1 = load_a_frag(aptr1, 0, hi);
    v16h bf[4];
    load_b4_f32<N>(bptr, bf);
#pragma unroll
    for (int j = 0; j < 4; ++j) {
      acc0[j] = wmma16(a0, bf[j], acc0[j]);
      acc1[j] = wmma16(a1, bf[j], acc1[j]);
    }
    aptr0 += 32;
    aptr1 += 32;
    bptr += (size_t)32 * N;
  }
#pragma unroll
  for (int j = 0; j < 4; ++j) {
#pragma unroll
    for (int v = 0; v < 8; ++v) {
      int row0 = mbase + v + 8 * hi;
      int row1 = row0 + 16;
      int col = n0 + j * 16 + lm;
      float v0 = acc0[j][v], v1 = acc1[j][v];
      if (residual) {
        v0 += residual[(size_t)row0 * N + col];
        v1 += residual[(size_t)row1 * N + col];
      }
      C[(size_t)row0 * N + col] = v0;
      C[(size_t)row1 * N + col] = v1;
    }
  }
}

// ---------------------------------------------------------------------------
// RoPE: split qkv, rotate q/k, emit head-major f16 Q/K and d-major f16 V^T
// grid = (T, HEADS), block = 64 (= HD/2)
// ---------------------------------------------------------------------------
__global__ void rope_kernel(const float* __restrict__ QKV,
                            const int* __restrict__ POS,
                            _Float16* __restrict__ Qh,
                            _Float16* __restrict__ Kh,
                            _Float16* __restrict__ VT) {
  int t = blockIdx.x, h = blockIdx.y, j = threadIdx.x;
  float p = (float)POS[t];
  float inv = __expf(-(float)j * (logf(1.0e6f) / 64.0f));
  float fr = p * inv;
  float c = cosf(fr), s = sinf(fr);
  const float* base = QKV + (size_t)t * (3 * HID_) + h * HD_;
  float q1 = base[j], q2 = base[j + 64];
  float k1 = base[HID_ + j], k2 = base[HID_ + j + 64];
  float v1 = base[2 * HID_ + j], v2 = base[2 * HID_ + j + 64];
  size_t qk = (size_t)(h * T_ + t) * HD_;
  Qh[qk + j]      = (_Float16)(q1 * c - q2 * s);
  Qh[qk + j + 64] = (_Float16)(q2 * c + q1 * s);
  Kh[qk + j]      = (_Float16)(k1 * c - k2 * s);
  Kh[qk + j + 64] = (_Float16)(k2 * c + k1 * s);
  VT[(size_t)(h * HD_ + j) * T_ + t]      = (_Float16)v1;
  VT[(size_t)(h * HD_ + j + 64) * T_ + t] = (_Float16)v2;
}

// ---------------------------------------------------------------------------
// Flash attention, one wave per (query-tile of 16, head).
// grid = (T/16, HEADS), block = 32
// ---------------------------------------------------------------------------
__global__ void __launch_bounds__(32, 1)
attn_kernel(const _Float16* __restrict__ Q,
            const _Float16* __restrict__ Km,
            const _Float16* __restrict__ VT,
            _Float16* __restrict__ Aout) {
  int mbase = blockIdx.x * 16;
  int head = blockIdx.y;
  int lane = threadIdx.x, lm = lane & 15, hi = lane >> 4;
  __shared__ _Float16 Pt[16][32];

  const _Float16* qrow = Q + (size_t)(head * T_ + mbase + lm) * HD_;
  v16h qa[4];
#pragma unroll
  for (int s4 = 0; s4 < 4; ++s4) qa[s4] = load_a_frag(qrow, s4 * 32, hi);

  v8f o[8];
#pragma unroll
  for (int j = 0; j < 8; ++j)
#pragma unroll
    for (int v = 0; v < 8; ++v) o[j][v] = 0.0f;
  float mrow[8], lrow[8], corrv[8];
#pragma unroll
  for (int v = 0; v < 8; ++v) { mrow[v] = -3.0e38f; lrow[v] = 0.0f; }

  int tmax = mbase + 15;
  for (int sb = 0; sb <= tmax; sb += 32) {
    const _Float16* krow0 = Km + (size_t)(head * T_ + sb + lm) * HD_;
    const _Float16* krow1 = krow0 + 16 * HD_;
    v8f s0, s1;
#pragma unroll
    for (int v = 0; v < 8; ++v) { s0[v] = 0.0f; s1[v] = 0.0f; }
#pragma unroll
    for (int s4 = 0; s4 < 4; ++s4) {
      int kk = s4 * 32;
      v16h b0 = load_a_frag(krow0, kk, hi);  // B = K^T -> A-pattern load of K
      s0 = wmma16(qa[s4], b0, s0);
      v16h b1 = load_a_frag(krow1, kk, hi);
      s1 = wmma16(qa[s4], b1, s1);
    }
    // scale + causal mask + online softmax (row = v + 8*hi, col = lane%16)
#pragma unroll
    for (int v = 0; v < 8; ++v) {
      int trow = mbase + v + 8 * hi;
      float x0 = s0[v] * SCALE_;
      float x1 = s1[v] * SCALE_;
      if (sb + lm > trow) x0 = -3.0e38f;
      if (sb + 16 + lm > trow) x1 = -3.0e38f;
      float mx = fmaxf(x0, x1);
#pragma unroll
      for (int d = 1; d < 16; d <<= 1) mx = fmaxf(mx, __shfl_xor(mx, d, 32));
      float mn = fmaxf(mrow[v], mx);
      float corr = __expf(mrow[v] - mn);
      float p0 = __expf(x0 - mn);
      float p1 = __expf(x1 - mn);
      float rs = p0 + p1;
#pragma unroll
      for (int d = 1; d < 16; d <<= 1) rs += __shfl_xor(rs, d, 32);
      lrow[v] = lrow[v] * corr + rs;
      mrow[v] = mn;
      corrv[v] = corr;
      s0[v] = p0;
      s1[v] = p1;
    }
#pragma unroll
    for (int j = 0; j < 8; ++j)
#pragma unroll
      for (int v = 0; v < 8; ++v) o[j][v] *= corrv[v];
    // transpose P (C-layout) -> A-fragment via LDS
#pragma unroll
    for (int v = 0; v < 8; ++v) {
      Pt[v + 8 * hi][lm] = (_Float16)s0[v];
      Pt[v + 8 * hi][16 + lm] = (_Float16)s1[v];
    }
    __syncthreads();
    v16h pa;
    {
      v8h lo = *(const v8h*)&Pt[lm][hi * 8];
      v8h hh = *(const v8h*)&Pt[lm][16 + hi * 8];
#pragma unroll
      for (int i = 0; i < 8; ++i) { pa[i] = lo[i]; pa[8 + i] = hh[i]; }
    }
    __syncthreads();
#pragma unroll
    for (int j = 0; j < 8; ++j) {
      const _Float16* vrow = VT + (size_t)(head * HD_ + j * 16 + lm) * T_;
      v16h bv;
      {
        v8h lo = *(const v8h*)(vrow + sb + hi * 8);
        v8h hh = *(const v8h*)(vrow + sb + 16 + hi * 8);
#pragma unroll
        for (int i = 0; i < 8; ++i) { bv[i] = lo[i]; bv[8 + i] = hh[i]; }
      }
      o[j] = wmma16(pa, bv, o[j]);
    }
  }
#pragma unroll
  for (int j = 0; j < 8; ++j)
#pragma unroll
    for (int v = 0; v < 8; ++v) {
      int trow = mbase + v + 8 * hi;
      Aout[(size_t)trow * HID_ + head * HD_ + j * 16 + lm] =
          (_Float16)(o[j][v] / lrow[v]);
    }
}

// ---------------------------------------------------------------------------
// Router: logits, softmax top-4, histogram/scan/scatter
// ---------------------------------------------------------------------------
__global__ void gate_kernel(const _Float16* __restrict__ H,
                            const float* __restrict__ GW,
                            float* __restrict__ L) {
  int t = blockIdx.x, e = threadIdx.x;
  if (e >= E_) return;
  float acc = 0.0f;
  for (int k = 0; k < HID_; ++k)
    acc += (float)H[(size_t)t * HID_ + k] * GW[(size_t)k * E_ + e];
  L[(size_t)t * E_ + e] = acc;
}

__global__ void topk_kernel(const float* __restrict__ L,
                            float* __restrict__ TW, int* __restrict__ TI) {
  int t = blockIdx.x * 64 + threadIdx.x;
  if (t >= T_) return;
  float lg[E_];
  float m = -3.0e38f;
  for (int e = 0; e < E_; ++e) { lg[e] = L[(size_t)t * E_ + e]; m = fmaxf(m, lg[e]); }
  float sum = 0.0f;
  for (int e = 0; e < E_; ++e) sum += __expf(lg[e] - m);
  float rsum = 1.0f / sum;
  for (int k = 0; k < TOPK_; ++k) {
    int bj = 0;
    float bv = -3.0e38f;
    for (int e = 0; e < E_; ++e)
      if (lg[e] > bv) { bv = lg[e]; bj = e; }
    TW[t * TOPK_ + k] = __expf(bv - m) * rsum;
    TI[t * TOPK_ + k] = bj;
    lg[bj] = -3.0e38f;
  }
}

__global__ void zero_cnt_kernel(int* cnt, int* fill) {
  int i = threadIdx.x;
  if (i < E_) { cnt[i] = 0; fill[i] = 0; }
}

__global__ void hist_kernel(const int* __restrict__ TI, int* cnt) {
  int p = blockIdx.x * 256 + threadIdx.x;
  if (p < NPAIR_) atomicAdd(&cnt[TI[p]], 1);
}

__global__ void scan_kernel(const int* __restrict__ cnt, int* off) {
  if (threadIdx.x == 0) {
    int s = 0;
    for (int e = 0; e < E_; ++e) { off[e] = s; s += cnt[e]; }
    off[E_] = s;
  }
}

__global__ void scatter_kernel(const int* __restrict__ TI,
                               const int* __restrict__ off, int* fill,
                               int* tok, int* pair_of) {
  int t = blockIdx.x * 64 + threadIdx.x;
  if (t >= T_) return;
  for (int k = 0; k < TOPK_; ++k) {
    int e = TI[t * TOPK_ + k];
    int pos = atomicAdd(&fill[e], 1);
    int pair = off[e] + pos;
    tok[pair] = t;
    pair_of[t * TOPK_ + k] = pair;
  }
}

// ---------------------------------------------------------------------------
// Grouped expert GEMM 1: SU[pair, MI] = silu(g)*u, g/u from h @ gu_w[e]
// grid = (64, MI/64, E), one wave per block, indirect A rows via tok list.
// ---------------------------------------------------------------------------
__global__ void __launch_bounds__(32, 1)
gemm_gu_expert_kernel(const _Float16* __restrict__ H,
                      const float* __restrict__ GUW,
                      const int* __restrict__ off,
                      const int* __restrict__ cnt,
                      const int* __restrict__ tok,
                      _Float16* __restrict__ SU) {
  int z = blockIdx.z;
  int count = cnt[z];
  int mbase = blockIdx.x * 16;
  if (mbase >= count) return;
  int lane = threadIdx.x, lm = lane & 15, hi = lane >> 4;
  int n0 = blockIdx.y * 64;
  int offz = off[z];
  int r = mbase + lm;
  int ridx = tok[offz + (r < count ? r : count - 1)];
  const _Float16* aptr = H + (size_t)ridx * HID_;
  const float* bptr = GUW + (size_t)z * ((size_t)HID_ * 2 * MI_) +
                      (size_t)(hi * 8) * (2 * MI_) + n0 + lm;

  v8f ag[4], au[4];
#pragma unroll
  for (int j = 0; j < 4; ++j)
#pragma unroll
    for (int v = 0; v < 8; ++v) { ag[j][v] = 0.0f; au[j][v] = 0.0f; }
  for (int kk = 0; kk < HID_; kk += 32) {
    v16h a = load_a_frag(aptr, 0, hi);
    v16h bf[4];
    load_b4_f32<2 * MI_>(bptr, bf);
#pragma unroll
    for (int j = 0; j < 4; ++j) ag[j] = wmma16(a, bf[j], ag[j]);
    load_b4_f32<2 * MI_>(bptr + MI_, bf);
#pragma unroll
    for (int j = 0; j < 4; ++j) au[j] = wmma16(a, bf[j], au[j]);
    aptr += 32;
    bptr += (size_t)32 * (2 * MI_);
  }
#pragma unroll
  for (int j = 0; j < 4; ++j)
#pragma unroll
    for (int v = 0; v < 8; ++v) {
      int rr = mbase + v + 8 * hi;
      if (rr < count) {
        int col = n0 + j * 16 + lm;
        SU[(size_t)(offz + rr) * MI_ + col] =
            (_Float16)(silu_f(ag[j][v]) * au[j][v]);
      }
    }
}

// ---------------------------------------------------------------------------
// Grouped expert GEMM 2: D[pair, HID] = SU[pair, MI] @ down_w[e]
// grid = (64, HID/64, E)
// ---------------------------------------------------------------------------
__global__ void __launch_bounds__(32, 1)
gemm_down_expert_kernel(const _Float16* __restrict__ SU,
                        const float* __restrict__ DW,
                        const int* __restrict__ off,
                        const int* __restrict__ cnt,
                        float* __restrict__ D) {
  int z = blockIdx.z;
  int count = cnt[z];
  int mbase = blockIdx.x * 16;
  if (mbase >= count) return;
  int lane = threadIdx.x, lm = lane & 15, hi = lane >> 4;
  int n0 = blockIdx.y * 64;
  int offz = off[z];
  int r = mbase + lm;
  const _Float16* aptr =
      SU + (size_t)(offz + (r < count ? r : count - 1)) * MI_;
  const float* bptr = DW + (size_t)z * ((size_t)MI_ * HID_) +
                      (size_t)(hi * 8) * HID_ + n0 + lm;

  v8f acc[4];
#pragma unroll
  for (int j = 0; j < 4; ++j)
#pragma unroll
    for (int v = 0; v < 8; ++v) acc[j][v] = 0.0f;
  for (int kk = 0; kk < MI_; kk += 32) {
    v16h a = load_a_frag(aptr, 0, hi);
    v16h bf[4];
    load_b4_f32<HID_>(bptr, bf);
#pragma unroll
    for (int j = 0; j < 4; ++j) acc[j] = wmma16(a, bf[j], acc[j]);
    aptr += 32;
    bptr += (size_t)32 * HID_;
  }
#pragma unroll
  for (int j = 0; j < 4; ++j)
#pragma unroll
    for (int v = 0; v < 8; ++v) {
      int rr = mbase + v + 8 * hi;
      if (rr < count)
        D[(size_t)(offz + rr) * HID_ + n0 + j * 16 + lm] = acc[j][v];
    }
}

// ---------------------------------------------------------------------------
// Shared expert gate/up with fused silu: SU[T, SI]
// grid = (T/16, SI/64)
// ---------------------------------------------------------------------------
__global__ void __launch_bounds__(32, 1)
gemm_gu_shared_kernel(const _Float16* __restrict__ H,
                      const float* __restrict__ GUW,
                      _Float16* __restrict__ SU) {
  int lane = threadIdx.x, lm = lane & 15, hi = lane >> 4;
  int mbase = blockIdx.x * 16;
  int n0 = blockIdx.y * 64;
  const _Float16* aptr = H + (size_t)(mbase + lm) * HID_;
  const float* bptr = GUW + (size_t)(hi * 8) * (2 * SI_) + n0 + lm;

  v8f ag[4], au[4];
#pragma unroll
  for (int j = 0; j < 4; ++j)
#pragma unroll
    for (int v = 0; v < 8; ++v) { ag[j][v] = 0.0f; au[j][v] = 0.0f; }
  for (int kk = 0; kk < HID_; kk += 32) {
    v16h a = load_a_frag(aptr, 0, hi);
    v16h bf[4];
    load_b4_f32<2 * SI_>(bptr, bf);
#pragma unroll
    for (int j = 0; j < 4; ++j) ag[j] = wmma16(a, bf[j], ag[j]);
    load_b4_f32<2 * SI_>(bptr + SI_, bf);
#pragma unroll
    for (int j = 0; j < 4; ++j) au[j] = wmma16(a, bf[j], au[j]);
    aptr += 32;
    bptr += (size_t)32 * (2 * SI_);
  }
#pragma unroll
  for (int j = 0; j < 4; ++j)
#pragma unroll
    for (int v = 0; v < 8; ++v) {
      int rr = mbase + v + 8 * hi;
      int col = n0 + j * 16 + lm;
      SU[(size_t)rr * SI_ + col] = (_Float16)(silu_f(ag[j][v]) * au[j][v]);
    }
}

__global__ void sgate_kernel(const _Float16* __restrict__ H,
                             const float* __restrict__ GW,
                             float* __restrict__ SG) {
  int t = blockIdx.x, lane = threadIdx.x;
  float acc = 0.0f;
  for (int k = lane; k < HID_; k += 32)
    acc += (float)H[(size_t)t * HID_ + k] * GW[k];
#pragma unroll
  for (int d = 1; d < 32; d <<= 1) acc += __shfl_xor(acc, d, 32);
  if (lane == 0) SG[t] = 1.0f / (1.0f + __expf(-acc));
}

// ---------------------------------------------------------------------------
// Combine: out = x1 + sigmoid_gate*shared + sum_k w_k * D[pair_k]
// grid = (T, HID/256), block 256
// ---------------------------------------------------------------------------
__global__ void combine_kernel(const float* __restrict__ X1,
                               const float* __restrict__ SG,
                               const float* __restrict__ SHOUT,
                               const float* __restrict__ TW,
                               const int* __restrict__ PAIR,
                               const float* __restrict__ D,
                               float* __restrict__ OUT) {
  int t = blockIdx.x;
  int c = blockIdx.y * 256 + threadIdx.x;
  size_t i = (size_t)t * HID_ + c;
  float v = X1[i] + SG[t] * SHOUT[i];
#pragma unroll
  for (int k = 0; k < TOPK_; ++k)
    v += TW[t * TOPK_ + k] * D[(size_t)PAIR[t * TOPK_ + k] * HID_ + c];
  OUT[i] = v;
}

// ---------------------------------------------------------------------------
// Host-side launch
// ---------------------------------------------------------------------------
static inline size_t align_up(size_t x) { return (x + 255) & ~(size_t)255; }

extern "C" void kernel_launch(void* const* d_in, const int* in_sizes, int n_in,
                              void* d_out, int out_size, void* d_ws, size_t ws_size,
                              hipStream_t stream) {
  const float* x        = (const float*)d_in[0];
  const float* ln1_w    = (const float*)d_in[1];
  const float* qkv_w    = (const float*)d_in[2];
  const float* qkv_b    = (const float*)d_in[3];
  const float* o_w      = (const float*)d_in[4];
  const float* ln2_w    = (const float*)d_in[5];
  const float* gate_w   = (const float*)d_in[6];
  const float* e_gu_w   = (const float*)d_in[7];
  const float* e_dn_w   = (const float*)d_in[8];
  const float* s_gu_w   = (const float*)d_in[9];
  const float* s_dn_w   = (const float*)d_in[10];
  const float* s_gate_w = (const float*)d_in[11];
  const int*   pos      = (const int*)d_in[12];
  float* out = (float*)d_out;

  char* ws = (char*)d_ws;
  size_t o = 0;
  auto alloc = [&](size_t bytes) {
    char* p = ws + o;
    o = align_up(o + bytes);
    return p;
  };
  _Float16* h1f16  = (_Float16*)alloc((size_t)T_ * HID_ * 2);
  float*    qkv    = (float*)   alloc((size_t)T_ * 3 * HID_ * 4);
  _Float16* qf16   = (_Float16*)alloc((size_t)T_ * HID_ * 2);
  _Float16* kf16   = (_Float16*)alloc((size_t)T_ * HID_ * 2);
  _Float16* vT16   = (_Float16*)alloc((size_t)T_ * HID_ * 2);
  _Float16* attn16 = (_Float16*)alloc((size_t)T_ * HID_ * 2);
  float*    x1     = (float*)   alloc((size_t)T_ * HID_ * 4);
  _Float16* h2f16  = (_Float16*)alloc((size_t)T_ * HID_ * 2);
  float*    logits = (float*)   alloc((size_t)T_ * E_ * 4);
  float*    topw   = (float*)   alloc((size_t)T_ * TOPK_ * 4);
  int*      topi   = (int*)     alloc((size_t)T_ * TOPK_ * 4);
  int*      cnt    = (int*)     alloc((size_t)E_ * 4);
  int*      off    = (int*)     alloc((size_t)(E_ + 1) * 4);
  int*      fill   = (int*)     alloc((size_t)E_ * 4);
  int*      tok    = (int*)     alloc((size_t)NPAIR_ * 4);
  int*      pairof = (int*)     alloc((size_t)NPAIR_ * 4);
  _Float16* su_e   = (_Float16*)alloc((size_t)NPAIR_ * MI_ * 2);
  float*    Dbuf   = (float*)   alloc((size_t)NPAIR_ * HID_ * 4);
  _Float16* su_s   = (_Float16*)alloc((size_t)T_ * SI_ * 2);
  float*    shout  = (float*)   alloc((size_t)T_ * HID_ * 4);
  float*    sg     = (float*)   alloc((size_t)T_ * 4);
  (void)ws_size; (void)in_sizes; (void)n_in; (void)out_size;

  // ---- attention block ----
  rmsnorm_kernel<<<T_, 256, 0, stream>>>(x, ln1_w, h1f16);
  gemm_wmma_kernel<3 * HID_, HID_><<<dim3(T_ / 32, (3 * HID_) / 64), 32, 0, stream>>>(
      h1f16, HID_, qkv_w, qkv_b, qkv, nullptr);
  rope_kernel<<<dim3(T_, HEADS_), 64, 0, stream>>>(qkv, pos, qf16, kf16, vT16);
  attn_kernel<<<dim3(T_ / 16, HEADS_), 32, 0, stream>>>(qf16, kf16, vT16, attn16);
  gemm_wmma_kernel<HID_, HID_><<<dim3(T_ / 32, HID_ / 64), 32, 0, stream>>>(
      attn16, HID_, o_w, nullptr, x1, x);  // +residual x

  // ---- MoE block ----
  rmsnorm_kernel<<<T_, 256, 0, stream>>>(x1, ln2_w, h2f16);
  gate_kernel<<<T_, 64, 0, stream>>>(h2f16, gate_w, logits);
  topk_kernel<<<T_ / 64, 64, 0, stream>>>(logits, topw, topi);
  zero_cnt_kernel<<<1, 64, 0, stream>>>(cnt, fill);
  hist_kernel<<<NPAIR_ / 256, 256, 0, stream>>>(topi, cnt);
  scan_kernel<<<1, 32, 0, stream>>>(cnt, off);
  scatter_kernel<<<T_ / 64, 64, 0, stream>>>(topi, off, fill, tok, pairof);

  gemm_gu_expert_kernel<<<dim3(T_ / 16, MI_ / 64, E_), 32, 0, stream>>>(
      h2f16, e_gu_w, off, cnt, tok, su_e);
  gemm_down_expert_kernel<<<dim3(T_ / 16, HID_ / 64, E_), 32, 0, stream>>>(
      su_e, e_dn_w, off, cnt, Dbuf);

  gemm_gu_shared_kernel<<<dim3(T_ / 16, SI_ / 64), 32, 0, stream>>>(
      h2f16, s_gu_w, su_s);
  gemm_wmma_kernel<HID_, SI_><<<dim3(T_ / 32, HID_ / 64), 32, 0, stream>>>(
      su_s, SI_, s_dn_w, nullptr, shout, nullptr);
  sgate_kernel<<<T_, 32, 0, stream>>>(h2f16, s_gate_w, sg);

  combine_kernel<<<dim3(T_, HID_ / 256), 256, 0, stream>>>(
      x1, sg, shout, topw, pairof, Dbuf, out);
}
